// ACT_69724499083282
// MI455X (gfx1250) — matrix operand order repaired and checked by reading, not compiled, not run on hardware
//
#include <hip/hip_runtime.h>
#include <hip/hip_bf16.h>
#include <cstddef>
#include <cstdint>

// ---------------------------------------------------------------------------
// ACT ponder network, restructured:
//   C_x = x @ W_x^T + b_hidden               (once,   8.6 GF)
//   loop t: s = bf16(C_x + s@W_s^T [+Wc])    (12x,   17.2 GF each, WMMA bf16)
//           online ACT halting -> acc_hid += w_t * s
//   out  = bf16(acc_hid) @ W_o^T + b_out     (once,   8.6 GF, direct to d_out)
// Weights (16MB bf16) stay resident in the 192MB L2 -> compute bound.
// GEMM core: v_wmma_f32_16x16x32_bf16, K-tile=64 (16 WMMAs per barrier pair),
// global->register->LDS double-buffered pipeline so global loads for tile k+1
// drain behind the WMMAs of tile k.
// ---------------------------------------------------------------------------

#define B_DIM     2048
#define D_IN      1024
#define H_DIM     2048
#define D_OUT     1024
#define MAX_STEPS 12
#define ACT_EPS   0.01f

typedef __bf16 bfloat;
typedef bfloat v8bf  __attribute__((ext_vector_type(8)));
typedef bfloat v16bf __attribute__((ext_vector_type(16)));
typedef float  v8f   __attribute__((ext_vector_type(8)));

#define TILE_M 128
#define TILE_N 128
#define TILE_K 64
#define LDS_STRIDE 80   // 64 + 16 pad: row pitch 160B stays 16B-aligned for ds_load_b128

// C[M,N] = A[M,K] * Bw[N,K]^T  (+ biasN[col] + crow[row,col] + wcN[col]),
// optional bf16 copy of the result (next-step A operand).
__global__ __launch_bounds__(256) void wmma_gemm_nt(
    const bfloat* __restrict__ A,
    const bfloat* __restrict__ Bw,
    float* __restrict__ C,
    int M, int N, int K,
    const float* __restrict__ biasN,
    const float* __restrict__ crow,
    const float* __restrict__ wcN,
    bfloat* __restrict__ bfOut)
{
    __shared__ bfloat As[TILE_M * LDS_STRIDE];
    __shared__ bfloat Bs[TILE_N * LDS_STRIDE];

    const int tid  = threadIdx.x;
    const int lane = tid & 31;
    const int wid  = tid >> 5;      // 8 waves
    const int wm   = wid >> 1;      // 0..3 : 32 rows each
    const int wn   = wid & 1;       // 0..1 : 64 cols each
    const int hs   = lane >> 4;     // wave32 half-select (ISA 7.12.2 layouts)
    const int l16  = lane & 15;

    const int m0 = blockIdx.y * TILE_M;
    const int n0 = blockIdx.x * TILE_N;

    v8f acc[2][4];
    #pragma unroll
    for (int i = 0; i < 2; ++i)
        #pragma unroll
        for (int j = 0; j < 4; ++j)
            #pragma unroll
            for (int e = 0; e < 8; ++e)
                acc[i][j][e] = 0.0f;

    const int ldRow = tid >> 3;        // 0..31
    const int ldCol = (tid & 7) * 8;   // 0..56 step 8  (16B-aligned b128 loads)

    // register-resident staging for the *next* K-tile (global double buffer)
    v8bf ra[4], rb[4];
    #pragma unroll
    for (int p = 0; p < 4; ++p) {
        const int r = ldRow + p * 32;
        ra[p] = *(const v8bf*)(&A[(size_t)(m0 + r) * K + ldCol]);
        rb[p] = *(const v8bf*)(&Bw[(size_t)(n0 + r) * K + ldCol]);
    }

    for (int kt = 0; kt < K; kt += TILE_K) {
        // drain staged tile into LDS (loads have had a full compute phase to land)
        #pragma unroll
        for (int p = 0; p < 4; ++p) {
            const int r = ldRow + p * 32;
            *(v8bf*)(&As[r * LDS_STRIDE + ldCol]) = ra[p];
            *(v8bf*)(&Bs[r * LDS_STRIDE + ldCol]) = rb[p];
        }
        __syncthreads();

        // issue next tile's global loads: they overlap with the 16 WMMAs below
        if (kt + TILE_K < K) {
            const int kn = kt + TILE_K;
            #pragma unroll
            for (int p = 0; p < 4; ++p) {
                const int r = ldRow + p * 32;
                ra[p] = *(const v8bf*)(&A[(size_t)(m0 + r) * K + kn + ldCol]);
                rb[p] = *(const v8bf*)(&Bw[(size_t)(n0 + r) * K + kn + ldCol]);
            }
            if (kn + TILE_K < K) {   // warm L2/WGP$ two tiles ahead
                __builtin_prefetch(&A[(size_t)(m0 + ldRow) * K + kn + TILE_K + ldCol], 0, 1);
                __builtin_prefetch(&Bw[(size_t)(n0 + ldRow) * K + kn + TILE_K + ldCol], 0, 1);
            }
        }

        #pragma unroll
        for (int kk = 0; kk < TILE_K; kk += 32) {
            // A fragment (16x32 bf16): lane m=l16; elems 0-7 = K[hs*8..], 8-15 = K[16+hs*8..]
            v16bf afrag[2];
            #pragma unroll
            for (int fm = 0; fm < 2; ++fm) {
                const bfloat* base = &As[(wm * 32 + fm * 16 + l16) * LDS_STRIDE + kk];
                v8bf lo = *(const v8bf*)(base + hs * 8);
                v8bf hi = *(const v8bf*)(base + 16 + hs * 8);
                afrag[fm] = __builtin_shufflevector(lo, hi,
                    0,1,2,3,4,5,6,7,8,9,10,11,12,13,14,15);
            }
            // B fragment (32x16 bf16): lane n=l16; elems 0-7 = K[hs*8..], 8-15 = K[16+hs*8..]
            v16bf bfrag[4];
            #pragma unroll
            for (int fn = 0; fn < 4; ++fn) {
                const bfloat* base = &Bs[(wn * 64 + fn * 16 + l16) * LDS_STRIDE + kk];
                v8bf lo = *(const v8bf*)(base + hs * 8);
                v8bf hi = *(const v8bf*)(base + 16 + hs * 8);
                bfrag[fn] = __builtin_shufflevector(lo, hi,
                    0,1,2,3,4,5,6,7,8,9,10,11,12,13,14,15);
            }

            #pragma unroll
            for (int fm = 0; fm < 2; ++fm)
                #pragma unroll
                for (int fn = 0; fn < 4; ++fn)
                    acc[fm][fn] = __builtin_amdgcn_wmma_f32_16x16x32_bf16(
                        false, afrag[fm], false, bfrag[fn],
                        (short)0, acc[fm][fn], false, false);
        }
        __syncthreads();
    }

    // Epilogue. C/D layout: elem r -> row = fm*16 + hs*8 + r, col = l16.
    #pragma unroll
    for (int fm = 0; fm < 2; ++fm) {
        #pragma unroll
        for (int fn = 0; fn < 4; ++fn) {
            const int col = n0 + wn * 64 + fn * 16 + l16;
            float extra = 0.0f;
            if (biasN) extra += biasN[col];
            if (wcN)   extra += wcN[col];
            #pragma unroll
            for (int r = 0; r < 8; ++r) {
                const int row = m0 + wm * 32 + fm * 16 + hs * 8 + r;
                const size_t idx = (size_t)row * N + col;
                float v = acc[fm][fn][r] + extra;
                if (crow) v += crow[idx];
                C[idx] = v;
                if (bfOut) bfOut[idx] = (bfloat)v;
            }
        }
    }
}

// One block per batch row: halting prob, online ACT weight, acc_hid += w * s_new.
__global__ __launch_bounds__(256) void halt_scale_kernel(
    const float* __restrict__ s_new, const float* __restrict__ w_halt,
    const float* __restrict__ b_halt, int t, int lastStep,
    float* __restrict__ cum, float* __restrict__ rem,
    int* __restrict__ nstep, int* __restrict__ halted,
    float* __restrict__ acc_hid)
{
    __shared__ float red[256];
    __shared__ float wsh;
    const int b   = blockIdx.x;
    const int tid = threadIdx.x;
    const float* row = s_new + (size_t)b * H_DIM;

    float vals[H_DIM / 256];
    float dot = 0.0f;
    #pragma unroll
    for (int e = 0; e < H_DIM / 256; ++e) {
        const int i = tid + e * 256;
        const float v = row[i];
        vals[e] = v;
        dot += v * w_halt[i];
    }
    red[tid] = dot;
    __syncthreads();
    for (int s = 128; s > 0; s >>= 1) {
        if (tid < s) red[tid] += red[tid + s];
        __syncthreads();
    }
    if (tid == 0) {
        const float z = red[0] + b_halt[0];
        const float p = 1.0f / (1.0f + __expf(-z));
        float w = 0.0f;
        if (!halted[b]) {
            const float c = cum[b];
            if (c + p >= 1.0f - ACT_EPS || lastStep) {
                w = 1.0f - c;           // remainder R
                rem[b] = w;
                nstep[b] = t;
                halted[b] = 1;
            } else {
                w = p;
                cum[b] = c + p;
            }
        }
        wsh = w;
    }
    __syncthreads();
    const float w = wsh;
    float* ah = acc_hid + (size_t)b * H_DIM;
    #pragma unroll
    for (int e = 0; e < H_DIM / 256; ++e) {
        const int i = tid + e * 256;
        ah[i] += w * vals[e];
    }
}

__global__ void split_whidden_kernel(const float* __restrict__ Wh,
                                     bfloat* __restrict__ Wx,
                                     bfloat* __restrict__ Ws)
{
    const size_t i = (size_t)blockIdx.x * 256 + threadIdx.x;
    const size_t total = (size_t)H_DIM * (D_IN + H_DIM);
    if (i >= total) return;
    const int row = (int)(i / (D_IN + H_DIM));
    const int col = (int)(i % (D_IN + H_DIM));
    const bfloat v = (bfloat)Wh[i];
    if (col < D_IN) Wx[(size_t)row * D_IN + col] = v;
    else            Ws[(size_t)row * H_DIM + (col - D_IN)] = v;
}

__global__ __launch_bounds__(256) void rowsum_kernel(
    const float* __restrict__ W, float* __restrict__ out, int rowlen, int stride)
{
    __shared__ float red[256];
    const int h = blockIdx.x, tid = threadIdx.x;
    const float* r = W + (size_t)h * stride;
    float s = 0.0f;
    for (int i = tid; i < rowlen; i += 256) s += r[i];
    red[tid] = s;
    __syncthreads();
    for (int k = 128; k > 0; k >>= 1) {
        if (tid < k) red[tid] += red[tid + k];
        __syncthreads();
    }
    if (tid == 0) out[h] = red[0];
}

__global__ void cvt_bf16_kernel(const float* __restrict__ in,
                                bfloat* __restrict__ out, size_t n)
{
    const size_t i = (size_t)blockIdx.x * 256 + threadIdx.x;
    if (i < n) out[i] = (bfloat)in[i];
}

__global__ void ponder_kernel(const int* __restrict__ nstep,
                              const float* __restrict__ rem,
                              float* __restrict__ dst)
{
    const int i = blockIdx.x * 256 + threadIdx.x;
    if (i < B_DIM) dst[i] = (float)(nstep[i] + 1) + rem[i];
}

extern "C" void kernel_launch(void* const* d_in, const int* in_sizes, int n_in,
                              void* d_out, int out_size, void* d_ws, size_t ws_size,
                              hipStream_t stream)
{
    (void)in_sizes; (void)n_in; (void)out_size; (void)ws_size;
    const float* x    = (const float*)d_in[0];
    const float* h0   = (const float*)d_in[1];
    const float* Wh   = (const float*)d_in[2];
    const float* bh   = (const float*)d_in[3];
    const float* whal = (const float*)d_in[4];
    const float* bhal = (const float*)d_in[5];
    const float* Wo   = (const float*)d_in[6];
    const float* bo   = (const float*)d_in[7];
    float* out = (float*)d_out;

    char* ws = (char*)d_ws;
    size_t off = 0;
    auto take = [&](size_t bytes) -> void* {
        void* p = (void*)(ws + off);
        off = (off + bytes + 255) & ~(size_t)255;
        return p;
    };

    bfloat* Whs_bf = (bfloat*)take((size_t)H_DIM * H_DIM * 2);
    bfloat* Wx_bf  = (bfloat*)take((size_t)H_DIM * D_IN * 2);
    bfloat* Wo_bf  = (bfloat*)take((size_t)D_OUT * H_DIM * 2);
    bfloat* x_bf   = (bfloat*)take((size_t)B_DIM * D_IN * 2);
    float*  Wc     = (float*) take((size_t)H_DIM * 4);
    float*  Cx     = (float*) take((size_t)B_DIM * H_DIM * 4);
    bfloat* sbf0   = (bfloat*)take((size_t)B_DIM * H_DIM * 2);
    bfloat* sbf1   = (bfloat*)take((size_t)B_DIM * H_DIM * 2);
    bfloat* hid_bf = (bfloat*)take((size_t)B_DIM * H_DIM * 2);
    float*  s_new  = (float*) take((size_t)B_DIM * H_DIM * 4);
    float*  acchid = (float*) take((size_t)B_DIM * H_DIM * 4);
    float*  cum    = (float*) take((size_t)B_DIM * 4);
    float*  rem    = (float*) take((size_t)B_DIM * 4);
    int*    nstep  = (int*)   take((size_t)B_DIM * 4);
    int*    halted = (int*)   take((size_t)B_DIM * 4);
    bfloat* sbf[2] = { sbf0, sbf1 };

    // --- one-time precompute (re-done every call: deterministic, no state) ---
    {
        const size_t n = (size_t)H_DIM * (D_IN + H_DIM);
        split_whidden_kernel<<<dim3((unsigned)((n + 255) / 256)), 256, 0, stream>>>(Wh, Wx_bf, Whs_bf);
    }
    rowsum_kernel<<<dim3(H_DIM), 256, 0, stream>>>(Wh, Wc, D_IN, D_IN + H_DIM);
    cvt_bf16_kernel<<<dim3((unsigned)(((size_t)D_OUT * H_DIM + 255) / 256)), 256, 0, stream>>>(
        Wo, Wo_bf, (size_t)D_OUT * H_DIM);
    cvt_bf16_kernel<<<dim3((unsigned)(((size_t)B_DIM * D_IN + 255) / 256)), 256, 0, stream>>>(
        x, x_bf, (size_t)B_DIM * D_IN);
    cvt_bf16_kernel<<<dim3((unsigned)(((size_t)B_DIM * H_DIM + 255) / 256)), 256, 0, stream>>>(
        h0, sbf[0], (size_t)B_DIM * H_DIM);

    hipMemsetAsync(acchid, 0, (size_t)B_DIM * H_DIM * 4, stream);
    hipMemsetAsync(cum,    0, (size_t)B_DIM * 4, stream);
    hipMemsetAsync(rem,    0, (size_t)B_DIM * 4, stream);
    hipMemsetAsync(nstep,  0, (size_t)B_DIM * 4, stream);
    hipMemsetAsync(halted, 0, (size_t)B_DIM * 4, stream);

    // C_x = x @ W_x^T + b_hidden   (step-invariant part of the concat GEMM)
    wmma_gemm_nt<<<dim3(H_DIM / TILE_N, B_DIM / TILE_M), 256, 0, stream>>>(
        x_bf, Wx_bf, Cx, B_DIM, H_DIM, D_IN, bh, nullptr, nullptr, nullptr);

    int cur = 0;
    for (int t = 0; t < MAX_STEPS; ++t) {
        // s_new = C_x + s @ W_s^T  (+ Wc at t==0 for the x+1 flag); bf16 copy for next step
        wmma_gemm_nt<<<dim3(H_DIM / TILE_N, B_DIM / TILE_M), 256, 0, stream>>>(
            sbf[cur], Whs_bf, s_new, B_DIM, H_DIM, H_DIM,
            nullptr, Cx, (t == 0) ? Wc : nullptr, sbf[cur ^ 1]);
        halt_scale_kernel<<<dim3(B_DIM), 256, 0, stream>>>(
            s_new, whal, bhal, t, (t == MAX_STEPS - 1) ? 1 : 0,
            cum, rem, nstep, halted, acchid);
        cur ^= 1;
    }

    // output = hidden @ W_o^T + b_out   (since sum of ACT weights == 1)
    cvt_bf16_kernel<<<dim3((unsigned)(((size_t)B_DIM * H_DIM + 255) / 256)), 256, 0, stream>>>(
        acchid, hid_bf, (size_t)B_DIM * H_DIM);
    wmma_gemm_nt<<<dim3(D_OUT / TILE_N, B_DIM / TILE_M), 256, 0, stream>>>(
        hid_bf, Wo_bf, out, B_DIM, D_OUT, H_DIM, bo, nullptr, nullptr, nullptr);

    // hidden and ponder outputs
    hipMemcpyAsync(out + (size_t)B_DIM * D_OUT, acchid,
                   (size_t)B_DIM * H_DIM * 4, hipMemcpyDeviceToDevice, stream);
    ponder_kernel<<<dim3(B_DIM / 256), 256, 0, stream>>>(
        nstep, rem, out + (size_t)B_DIM * D_OUT + (size_t)B_DIM * H_DIM);
}
